// MultiHeadCrossAttention_12068858102017
// MI455X (gfx1250) — matrix operand order repaired and checked by reading, not compiled, not run on hardware
//
#include <hip/hip_runtime.h>

// ---------------------------------------------------------------------------
// MultiHeadCrossAttention for MI455X (gfx1250, wave32, WMMA f16->f32)
// B=2, Tq=1024, Tkv=4096, C=1024, H=16, D=64
// Pipeline: cvt(x)->f16 ; Wt=transpose(W)->f16 ; Q=xWq (scaled 1/8) ;
//           K,V=xWkv ; flash-attention (online softmax) -> Y f16 ; out=Y@Wc f32
// ---------------------------------------------------------------------------

typedef _Float16 v16h __attribute__((ext_vector_type(16)));
typedef _Float16 v8h  __attribute__((ext_vector_type(8)));
typedef _Float16 v4h  __attribute__((ext_vector_type(4)));
typedef float    v8f  __attribute__((ext_vector_type(8)));

#define WMMA_F16(a, b, c) \
  __builtin_amdgcn_wmma_f32_16x16x32_f16(false, (a), false, (b), (short)0, (c), false, false)

// ---- VALU lane-butterfly within each 16-lane row (v_permlane16_b32) --------
// Avoids ds_bpermute round trips for the softmax row reductions.
template <unsigned SELLO, unsigned SELHI>
static __device__ __forceinline__ float permxor16(float v) {
  unsigned u = __builtin_bit_cast(unsigned, v);
  unsigned r = __builtin_amdgcn_permlane16(u, u, SELLO, SELHI, false, false);
  return __builtin_bit_cast(float, r);
}
static __device__ __forceinline__ float rowmax16(float v) {
  v = fmaxf(v, permxor16<0x67452301u, 0xEFCDAB89u>(v));  // xor 1
  v = fmaxf(v, permxor16<0x54761032u, 0xDCFE98BAu>(v));  // xor 2
  v = fmaxf(v, permxor16<0x32107654u, 0xBA98FEDCu>(v));  // xor 4
  v = fmaxf(v, permxor16<0xFEDCBA98u, 0x76543210u>(v));  // xor 8
  return v;
}
static __device__ __forceinline__ float rowsum16(float v) {
  v += permxor16<0x67452301u, 0xEFCDAB89u>(v);
  v += permxor16<0x54761032u, 0xDCFE98BAu>(v);
  v += permxor16<0x32107654u, 0xBA98FEDCu>(v);
  v += permxor16<0xFEDCBA98u, 0x76543210u>(v);
  return v;
}

// ---- fragment loaders (contiguous k = 16g + e order, same for A and B) ---------------
// A (16x32 f16): lane holds row m=lane%16; g=lane/16.
//   elements [0..7]  = row[16g .. 16g+7]
//   elements [8..15] = row[16g+8 .. 16g+15]
static __device__ __forceinline__ v16h ld_afrag(const _Float16* row, int g) {
  v8h lo = *(const v8h*)(row + 16 * g);
  v8h hi = *(const v8h*)(row + 16 * g + 8);
  v16h r;
#pragma unroll
  for (int i = 0; i < 8; ++i) { r[i] = lo[i]; r[i + 8] = hi[i]; }
  return r;
}
// B (32x16 f16): lane holds col n=lane%16; g=lane/16; element e <-> K=16g+e.
// caller passes pointer already offset by 16g along K.
static __device__ __forceinline__ v16h ld_bfrag(const _Float16* p) {
  v8h lo = *(const v8h*)(p);
  v8h hi = *(const v8h*)(p + 8);
  v16h r;
#pragma unroll
  for (int i = 0; i < 8; ++i) { r[i] = lo[i]; r[i + 8] = hi[i]; }
  return r;
}

// ---------------------------------------------------------------------------
// elementwise f32 -> f16 (4 elements / thread)
__global__ __launch_bounds__(256) void cvt_f32_f16_kernel(
    const float* __restrict__ in, _Float16* __restrict__ out, int n4) {
  int i = blockIdx.x * 256 + threadIdx.x;
  if (i < n4) {
    float4 v = ((const float4*)in)[i];
    v4h r; r[0] = (_Float16)v.x; r[1] = (_Float16)v.y;
    r[2] = (_Float16)v.z; r[3] = (_Float16)v.w;
    ((v4h*)out)[i] = r;
  }
}

// W [1024, N] f32  ->  Wt [N, 1024] f16   (32x32 LDS tile transpose)
__global__ __launch_bounds__(256) void transpose_cvt_kernel(
    const float* __restrict__ W, _Float16* __restrict__ Wt, int N) {
  __shared__ _Float16 tile[32][33];
  int n0 = blockIdx.x * 32, k0 = blockIdx.y * 32;
  int tx = threadIdx.x, ty = threadIdx.y;  // 32 x 8
#pragma unroll
  for (int i = 0; i < 4; ++i)
    tile[ty + 8 * i][tx] = (_Float16)W[(k0 + ty + 8 * i) * N + (n0 + tx)];
  __syncthreads();
#pragma unroll
  for (int i = 0; i < 4; ++i)
    Wt[(n0 + ty + 8 * i) * 1024 + (k0 + tx)] = tile[tx][ty + 8 * i];
}

// ---------------------------------------------------------------------------
// WMMA GEMM: C[M,N] = A[M,1024] @ Bt[N,1024]^T, K fixed 1024.
// WG = 256 thr = 8 waves (4x2), block tile 128x128, wave tile 32x64
// (2x4 wmma accum tiles, 8 wmma per 32-wide K step).
// MODE 0: outF = C (f32, [M,1024])                       (final projection)
// MODE 1: outH0[B,H,1024,64] = C * 0.125 (f16)           (Q, pre-scaled)
// MODE 2: cols<1024 -> outH0[B,H,4096,64]; else outH1    (K and V)
template <int MODE>
__global__ __launch_bounds__(256) void gemm_wmma_kernel(
    const _Float16* __restrict__ A, const _Float16* __restrict__ Bt,
    _Float16* __restrict__ outH0, _Float16* __restrict__ outH1,
    float* __restrict__ outF) {
  __shared__ _Float16 Als[128][40];
  __shared__ _Float16 Bls[128][40];
  const int tid = threadIdx.x;
  const int lane = tid & 31, wave = tid >> 5;
  const int wm = wave >> 1, wn = wave & 1;
  const int g = lane >> 4, ln = lane & 15;
  const int row0 = blockIdx.x * 128;
  const int col0 = blockIdx.y * 128;

  const int crow = tid >> 1, csub = tid & 1;  // 128 rows x 2 chunks of 16
  const _Float16* aSrc = A + (size_t)(row0 + crow) * 1024 + 16 * csub;
  const _Float16* bSrc = Bt + (size_t)(col0 + crow) * 1024 + 16 * csub;

  v8f acc[2][4] = {};
  for (int k0 = 0; k0 < 1024; k0 += 32) {
    __syncthreads();
    v8h a0 = *(const v8h*)(aSrc + k0);
    v8h a1 = *(const v8h*)(aSrc + k0 + 8);
    v8h b0 = *(const v8h*)(bSrc + k0);
    v8h b1 = *(const v8h*)(bSrc + k0 + 8);
    *(v8h*)&Als[crow][16 * csub] = a0;
    *(v8h*)&Als[crow][16 * csub + 8] = a1;
    *(v8h*)&Bls[crow][16 * csub] = b0;
    *(v8h*)&Bls[crow][16 * csub + 8] = b1;
    __syncthreads();

    v16h af[2], bf[4];
#pragma unroll
    for (int mi = 0; mi < 2; ++mi)
      af[mi] = ld_afrag(&Als[wm * 32 + mi * 16 + ln][0], g);
#pragma unroll
    for (int ni = 0; ni < 4; ++ni)
      bf[ni] = ld_bfrag(&Bls[wn * 64 + ni * 16 + ln][16 * g]);
#pragma unroll
    for (int mi = 0; mi < 2; ++mi)
#pragma unroll
      for (int ni = 0; ni < 4; ++ni)
        acc[mi][ni] = WMMA_F16(af[mi], bf[ni], acc[mi][ni]);
  }

#pragma unroll
  for (int mi = 0; mi < 2; ++mi)
#pragma unroll
    for (int ni = 0; ni < 4; ++ni)
#pragma unroll
      for (int r = 0; r < 8; ++r) {
        int R = row0 + wm * 32 + mi * 16 + r + 8 * g;
        int c = col0 + wn * 64 + ni * 16 + ln;
        float v = acc[mi][ni][r];
        if (MODE == 0) {
          outF[(size_t)R * 1024 + c] = v;
        } else if (MODE == 1) {           // Q: [B,H,Tq=1024,D=64], scale 1/8
          int b = R >> 10, t = R & 1023, h = c >> 6, d = c & 63;
          outH0[(((size_t)(b * 16 + h)) * 1024 + t) * 64 + d] =
              (_Float16)(v * 0.125f);
        } else {                          // KV: [B,H,Tkv=4096,D=64]
          int b = R >> 12, t = R & 4095;
          if (c < 1024) {
            int h = c >> 6, d = c & 63;
            outH0[(((size_t)(b * 16 + h)) * 4096 + t) * 64 + d] = (_Float16)v;
          } else {
            int c2 = c - 1024, h = c2 >> 6, d = c2 & 63;
            outH1[(((size_t)(b * 16 + h)) * 4096 + t) * 64 + d] = (_Float16)v;
          }
        }
      }
}

// ---------------------------------------------------------------------------
// Flash attention: grid (Tq/128, H, B), 256 thr = 8 waves; wave owns 16 q rows.
// kv block = 32; per block: 4 wmma (S) + softmax + 4 wmma (PV).
__global__ __launch_bounds__(256) void attention_kernel(
    const _Float16* __restrict__ Qh,   // [B,H,1024,64] pre-scaled by 1/8
    const _Float16* __restrict__ Kh,   // [B,H,4096,64]
    const _Float16* __restrict__ Vh,   // [B,H,4096,64]
    const unsigned char* __restrict__ qMask,   // [B,1024] bool
    const unsigned char* __restrict__ kvMask,  // [B,4096] bool
    _Float16* __restrict__ Yh) {       // [B,1024,1024]
  __shared__ _Float16 Kls[32][72];     // K block, row-major [kv][d]
  __shared__ _Float16 Vtls[64][40];    // V block transposed [d][kv]
  __shared__ _Float16 Pls[8][16][40];  // per-wave P (16 x 32)

  const int tid = threadIdx.x, lane = tid & 31, wave = tid >> 5;
  const int g = lane >> 4, ln = lane & 15;
  const int h = blockIdx.y, b = blockIdx.z;
  const int q0 = blockIdx.x * 128 + wave * 16;
  const size_t bh = (size_t)b * 16 + h;

  // Q fragments (loop invariant): 16 rows x 64, as 2 A-fragments (K=32 each)
  const _Float16* qrow = Qh + (bh * 1024 + q0 + ln) * 64;
  v16h qf[2];
  qf[0] = ld_afrag(qrow, g);
  qf[1] = ld_afrag(qrow + 32, g);

  bool qok[8];
#pragma unroll
  for (int r = 0; r < 8; ++r)
    qok[r] = qMask[(size_t)b * 1024 + q0 + r + 8 * g] != 0;

  v8f o[4] = {};
  float mrun[8], lrun[8];
#pragma unroll
  for (int r = 0; r < 8; ++r) { mrun[r] = -1e9f; lrun[r] = 0.0f; }

  const _Float16* kBase = Kh + bh * 4096 * 64;
  const _Float16* vBase = Vh + bh * 4096 * 64;
  const int lrow = tid >> 3, lsub = tid & 7;  // 32 rows x 8 chunks of 8 halves

  for (int kv0 = 0; kv0 < 4096; kv0 += 32) {
    __syncthreads();
    {
      v8h kx = *(const v8h*)(kBase + (kv0 + lrow) * 64 + 8 * lsub);
      v8h vx = *(const v8h*)(vBase + (kv0 + lrow) * 64 + 8 * lsub);
      *(v8h*)&Kls[lrow][8 * lsub] = kx;
#pragma unroll
      for (int j = 0; j < 8; ++j) Vtls[8 * lsub + j][lrow] = vx[j];
      if (kv0 + 32 < 4096) {  // prefetch next block (global_prefetch_b8)
        __builtin_prefetch(kBase + (kv0 + 32 + lrow) * 64 + 8 * lsub, 0, 1);
        __builtin_prefetch(vBase + (kv0 + 32 + lrow) * 64 + 8 * lsub, 0, 1);
      }
    }
    __syncthreads();

    bool kvb[2];
#pragma unroll
    for (int t = 0; t < 2; ++t)
      kvb[t] = kvMask[(size_t)b * 4096 + kv0 + 16 * t + ln] != 0;

    // S = Q @ K^T  (two 16x16 tiles, K-dim 64 split in two 32-chunks)
    v8f s[2];
#pragma unroll
    for (int t = 0; t < 2; ++t) {
      v8f st = {};
#pragma unroll
      for (int kc = 0; kc < 2; ++kc) {
        v16h bf = ld_bfrag(&Kls[16 * t + ln][32 * kc + 16 * g]);
        st = WMMA_F16(qf[kc], bf, st);
      }
      s[t] = st;
    }

    // mask + online softmax (row = r + 8g, reduce across 16 lanes of group)
    float p[2][8], scl[8];
#pragma unroll
    for (int r = 0; r < 8; ++r) {
      float s0 = (kvb[0] && qok[r]) ? s[0][r] : -1e9f;
      float s1 = (kvb[1] && qok[r]) ? s[1][r] : -1e9f;
      float mx = rowmax16(fmaxf(s0, s1));   // v_permlane16 butterfly (VALU)
      float mnew = fmaxf(mrun[r], mx);
      float p0 = __expf(s0 - mnew);
      float p1 = __expf(s1 - mnew);
      p[0][r] = p0; p[1][r] = p1;
      float sum = rowsum16(p0 + p1);        // v_permlane16 butterfly (VALU)
      float sc = __expf(mrun[r] - mnew);
      scl[r] = sc;
      lrun[r] = lrun[r] * sc + sum;
      mrun[r] = mnew;
    }
#pragma unroll
    for (int dt = 0; dt < 4; ++dt)
#pragma unroll
      for (int r = 0; r < 8; ++r) o[dt][r] *= scl[r];

    // transpose P through per-wave LDS: C-layout (col-in-lane) -> A-layout
#pragma unroll
    for (int t = 0; t < 2; ++t)
#pragma unroll
      for (int r = 0; r < 8; ++r)
        Pls[wave][r + 8 * g][16 * t + ln] = (_Float16)p[t][r];
    asm volatile("s_wait_dscnt 0" ::: "memory");  // wave-local RAW on LDS

    // O += P @ V   (4 output d-tiles of 16 cols, K=32)
    v16h pf = ld_afrag(&Pls[wave][ln][0], g);
#pragma unroll
    for (int dt = 0; dt < 4; ++dt) {
      v16h vf = ld_bfrag(&Vtls[16 * dt + ln][16 * g]);
      o[dt] = WMMA_F16(pf, vf, o[dt]);
    }
  }

  // normalize and write Y[b, t, h*64 + d] (f16)
#pragma unroll
  for (int r = 0; r < 8; ++r) lrun[r] = 1.0f / lrun[r];
#pragma unroll
  for (int dt = 0; dt < 4; ++dt)
#pragma unroll
    for (int r = 0; r < 8; ++r)
      Yh[((size_t)b * 1024 + q0 + r + 8 * g) * 1024 + h * 64 + 16 * dt + ln] =
          (_Float16)(o[dt][r] * lrun[r]);
}

// ---------------------------------------------------------------------------
extern "C" void kernel_launch(void* const* d_in, const int* in_sizes, int n_in,
                              void* d_out, int out_size, void* d_ws,
                              size_t ws_size, hipStream_t stream) {
  (void)in_sizes; (void)n_in; (void)out_size; (void)ws_size;
  const float* x_q  = (const float*)d_in[0];  // [2,1024,1024]
  const float* x_kv = (const float*)d_in[1];  // [2,4096,1024]
  const unsigned char* q_mask  = (const unsigned char*)d_in[2];  // bool [2,1024]
  const unsigned char* kv_mask = (const unsigned char*)d_in[3];  // bool [2,4096]
  const float* W_q  = (const float*)d_in[4];  // [1024,1024]
  const float* W_kv = (const float*)d_in[5];  // [1024,2048]
  const float* W_c  = (const float*)d_in[6];  // [1024,1024]
  float* out = (float*)d_out;                 // [2,1024,1024] f32

  _Float16* ws   = (_Float16*)d_ws;
  _Float16* xqh  = ws;                         // 2048*1024
  _Float16* xkvh = xqh  + (size_t)2048 * 1024; // 8192*1024
  _Float16* wqt  = xkvh + (size_t)8192 * 1024; // [1024,1024] = W_q^T
  _Float16* wkvt = wqt  + (size_t)1024 * 1024; // [2048,1024] = W_kv^T
  _Float16* wct  = wkvt + (size_t)2048 * 1024; // [1024,1024] = W_c^T
  _Float16* qh   = wct  + (size_t)1024 * 1024; // [2,16,1024,64]
  _Float16* kh   = qh   + (size_t)2048 * 1024; // [2,16,4096,64]
  _Float16* vh   = kh   + (size_t)8192 * 1024;
  _Float16* yh   = vh   + (size_t)8192 * 1024; // [2,1024,1024]

  // 1) convert activations to f16
  cvt_f32_f16_kernel<<<(2048 * 1024) / 4 / 256, 256, 0, stream>>>(x_q, xqh,
                                                                  2048 * 256);
  cvt_f32_f16_kernel<<<(8192 * 1024) / 4 / 256, 256, 0, stream>>>(x_kv, xkvh,
                                                                  8192 * 256);
  // 2) transpose+convert weights -> [N, K] f16
  transpose_cvt_kernel<<<dim3(32, 32), dim3(32, 8), 0, stream>>>(W_q, wqt, 1024);
  transpose_cvt_kernel<<<dim3(64, 32), dim3(32, 8), 0, stream>>>(W_kv, wkvt, 2048);
  transpose_cvt_kernel<<<dim3(32, 32), dim3(32, 8), 0, stream>>>(W_c, wct, 1024);
  // 3) Q = x_q @ W_q * 1/sqrt(64)  -> [B,H,Tq,D] f16
  gemm_wmma_kernel<1><<<dim3(16, 8), 256, 0, stream>>>(xqh, wqt, qh, nullptr,
                                                       nullptr);
  // 4) K,V = x_kv @ W_kv -> [B,H,Tkv,D] f16 each
  gemm_wmma_kernel<2><<<dim3(64, 16), 256, 0, stream>>>(xkvh, wkvt, kh, vh,
                                                        nullptr);
  // 5) flash attention -> Y f16 [B,Tq,C]
  attention_kernel<<<dim3(8, 16, 2), 256, 0, stream>>>(qh, kh, vh, q_mask,
                                                       kv_mask, yh);
  // 6) out = Y @ W_c -> f32
  gemm_wmma_kernel<0><<<dim3(16, 8), 256, 0, stream>>>(yh, wct, nullptr,
                                                       nullptr, out);
}
